// HSIC_84301618085995
// MI455X (gfx1250) — compile-verified, matrix operand order checked by reading, not run on hardware
//
#include <hip/hip_runtime.h>
#include <cstddef>
#include <cstdint>

// ---------------------------------------------------------------------------
// HSIC(X, Y), X,Y: [4096, 512] fp32.
//   hsic = [ S2 - (2/n)<rsX,rsY> + sX*sY/n^2 ] / (n-1)^2
// where Kx = exp(-0.5*(sq_i+sq_j-2*X X^T)), S2 = sum(Kx*Ky), rs = row sums.
// Single fused WMMA pass; bf16 hi/lo split for ~fp32-accurate Gram matrices.
// ---------------------------------------------------------------------------

#define HSIC_N 4096
#define HSIC_D 512
#define BM 128
#define BN 64
#define GRID_I (HSIC_N / BM)     // 32
#define GRID_J (HSIC_N / BN)     // 64
#define NBLK   (GRID_I * GRID_J) // 2048

typedef __attribute__((ext_vector_type(16))) __bf16 v16bf;
typedef __attribute__((ext_vector_type(8)))  float  v8f;

union FragU { v16bf v; uint4 q[2]; };

// 16-bit A/B fragment (16 rows x 32 K), ISA layout:
// lanes 0-15: row = lane,    K = [k0..k0+7] -> elems 0..7, [k0+16..k0+23] -> elems 8..15
// lanes 16-31: row = lane-16, K offset +8 (i.e. [k0+8..15], [k0+24..31])
__device__ __forceinline__ v16bf load_frag(const unsigned short* __restrict__ base,
                                           int row, int k0, int lane) {
  const int r  = row + (lane & 15);
  const int kb = k0 + ((lane >> 4) << 3);
  const unsigned short* p = base + (size_t)r * HSIC_D + kb;
  FragU f;
  f.q[0] = *reinterpret_cast<const uint4*>(p);       // 8 bf16, 16B aligned
  f.q[1] = *reinterpret_cast<const uint4*>(p + 16);  // next 8 bf16 (K+16)
  return f.v;
}

__device__ __forceinline__ v8f wmma_bf16(v16bf a, v16bf b, v8f c) {
  // (neg_a, A, neg_b, B, c_mod, C, reuse_a, reuse_b)
  return __builtin_amdgcn_wmma_f32_16x16x32_bf16(false, a, false, b, (short)0, c,
                                                 false, false);
}

// ---------------------------------------------------------------------------
// Kernel 1: fp32 -> (bf16 hi, bf16 lo) split + exact fp32 squared row norms.
// grid: (4096 rows, 2 inputs), block: 256
// ---------------------------------------------------------------------------
__global__ __launch_bounds__(256) void hsic_convert(
    const float* __restrict__ X, const float* __restrict__ Y,
    unsigned short* __restrict__ Xhi, unsigned short* __restrict__ Xlo,
    unsigned short* __restrict__ Yhi, unsigned short* __restrict__ Ylo,
    float* __restrict__ sqX, float* __restrict__ sqY) {
  const float* in; unsigned short* h; unsigned short* l; float* sq;
  if (blockIdx.y == 0) { in = X; h = Xhi; l = Xlo; sq = sqX; }
  else                 { in = Y; h = Yhi; l = Ylo; sq = sqY; }
  const int row = blockIdx.x;
  float s = 0.f;
  for (int c = threadIdx.x; c < HSIC_D; c += 256) {
    float x = in[(size_t)row * HSIC_D + c];
    unsigned u  = __float_as_uint(x);
    unsigned hb = (u + 0x7FFFu + ((u >> 16) & 1u)) & 0xFFFF0000u;  // RNE to bf16
    float hf = __uint_as_float(hb);
    float lof = x - hf;
    unsigned ul = __float_as_uint(lof);
    unsigned lb = (ul + 0x7FFFu + ((ul >> 16) & 1u)) & 0xFFFF0000u;
    h[(size_t)row * HSIC_D + c] = (unsigned short)(hb >> 16);
    l[(size_t)row * HSIC_D + c] = (unsigned short)(lb >> 16);
    s = fmaf(x, x, s);
  }
  __shared__ float red[256];
  red[threadIdx.x] = s;
  __syncthreads();
  for (int st = 128; st > 0; st >>= 1) {
    if (threadIdx.x < st) red[threadIdx.x] += red[threadIdx.x + st];
    __syncthreads();
  }
  if (threadIdx.x == 0) sq[row] = red[0];
}

// ---------------------------------------------------------------------------
// Kernel 2: fused Gram + RBF + partial reductions.
// Block tile 128x64; 8 waves in 4x2; each wave owns a 32x32 tile (2x2 WMMA).
// Per k-step (K=32): 24 v_wmma_f32_16x16x32_bf16 (hi/lo split x 2 matrices).
// ---------------------------------------------------------------------------
__global__ __launch_bounds__(256) void hsic_main(
    const unsigned short* __restrict__ Xhi, const unsigned short* __restrict__ Xlo,
    const unsigned short* __restrict__ Yhi, const unsigned short* __restrict__ Ylo,
    const float* __restrict__ sqX, const float* __restrict__ sqY,
    float* __restrict__ S2p, float* __restrict__ rsXp, float* __restrict__ rsYp) {
  const int lane = threadIdx.x & 31;
  const int wave = threadIdx.x >> 5;  // 0..7
  const int wm = wave & 3;            // i direction: 4 waves x 32 rows
  const int wn = wave >> 2;           // j direction: 2 waves x 32 cols
  const int bi = blockIdx.x, bj = blockIdx.y;
  const int i0 = bi * BM + wm * 32;
  const int j0 = bj * BN + wn * 32;

  __shared__ float rsx_s[BM];
  __shared__ float rsy_s[BM];
  __shared__ float s2_s[8];
  if (threadIdx.x < BM) { rsx_s[threadIdx.x] = 0.f; rsy_s[threadIdx.x] = 0.f; }
  __syncthreads();

  const v8f vzero = {0.f, 0.f, 0.f, 0.f, 0.f, 0.f, 0.f, 0.f};
  v8f gx[2][2], gy[2][2];
#pragma unroll
  for (int mi = 0; mi < 2; ++mi)
#pragma unroll
    for (int ni = 0; ni < 2; ++ni) { gx[mi][ni] = vzero; gy[mi][ni] = vzero; }

  for (int k0 = 0; k0 < HSIC_D; k0 += 32) {
    if (k0 + 32 < HSIC_D) {  // hint next K slice (global_prefetch_b8)
      __builtin_prefetch(Xhi + (size_t)(i0 + (lane & 15)) * HSIC_D + (k0 + 32), 0, 1);
      __builtin_prefetch(Yhi + (size_t)(j0 + (lane & 15)) * HSIC_D + (k0 + 32), 0, 1);
    }
    {  // X Gram: split product hi*hi + hi*lo + lo*hi
      v16bf ah[2], al[2], bh[2], bl[2];
#pragma unroll
      for (int t = 0; t < 2; ++t) {
        ah[t] = load_frag(Xhi, i0 + 16 * t, k0, lane);
        al[t] = load_frag(Xlo, i0 + 16 * t, k0, lane);
        bh[t] = load_frag(Xhi, j0 + 16 * t, k0, lane);
        bl[t] = load_frag(Xlo, j0 + 16 * t, k0, lane);
      }
#pragma unroll
      for (int mi = 0; mi < 2; ++mi)
#pragma unroll
        for (int ni = 0; ni < 2; ++ni) {
          gx[mi][ni] = wmma_bf16(ah[mi], bh[ni], gx[mi][ni]);
          gx[mi][ni] = wmma_bf16(ah[mi], bl[ni], gx[mi][ni]);
          gx[mi][ni] = wmma_bf16(al[mi], bh[ni], gx[mi][ni]);
        }
    }
    {  // Y Gram
      v16bf ah[2], al[2], bh[2], bl[2];
#pragma unroll
      for (int t = 0; t < 2; ++t) {
        ah[t] = load_frag(Yhi, i0 + 16 * t, k0, lane);
        al[t] = load_frag(Ylo, i0 + 16 * t, k0, lane);
        bh[t] = load_frag(Yhi, j0 + 16 * t, k0, lane);
        bl[t] = load_frag(Ylo, j0 + 16 * t, k0, lane);
      }
#pragma unroll
      for (int mi = 0; mi < 2; ++mi)
#pragma unroll
        for (int ni = 0; ni < 2; ++ni) {
          gy[mi][ni] = wmma_bf16(ah[mi], bh[ni], gy[mi][ni]);
          gy[mi][ni] = wmma_bf16(ah[mi], bl[ni], gy[mi][ni]);
          gy[mi][ni] = wmma_bf16(al[mi], bh[ni], gy[mi][ni]);
        }
    }
  }

  // Epilogue. C layout: VGPR v, lanes 0-15 -> (M=v, N=lane); lanes 16-31 -> (M=8+v).
  const int jn = lane & 15;
  const int hb8 = (lane >> 4) << 3;
  float s2l = 0.f;
#pragma unroll
  for (int mi = 0; mi < 2; ++mi) {
    float sqxi[8], sqyi[8];
#pragma unroll
    for (int v = 0; v < 8; ++v) {
      const int i = i0 + 16 * mi + hb8 + v;
      sqxi[v] = sqX[i]; sqyi[v] = sqY[i];
    }
    float rx[8], ry[8];
#pragma unroll
    for (int v = 0; v < 8; ++v) { rx[v] = 0.f; ry[v] = 0.f; }
#pragma unroll
    for (int ni = 0; ni < 2; ++ni) {
      const int j = j0 + 16 * ni + jn;
      const float sxj = sqX[j], syj = sqY[j];
#pragma unroll
      for (int v = 0; v < 8; ++v) {
        const float kx = __expf(-0.5f * (sqxi[v] + sxj - 2.f * gx[mi][ni][v]));
        const float ky = __expf(-0.5f * (sqyi[v] + syj - 2.f * gy[mi][ni][v]));
        s2l = fmaf(kx, ky, s2l);
        rx[v] += kx; ry[v] += ky;
      }
    }
    // Reduce row-sum partials across each 16-lane half (xor<16 stays in half).
#pragma unroll
    for (int v = 0; v < 8; ++v) {
      float r1 = rx[v], r2 = ry[v];
#pragma unroll
      for (int off = 1; off < 16; off <<= 1) {
        r1 += __shfl_xor(r1, off, 32);
        r2 += __shfl_xor(r2, off, 32);
      }
      if (jn == 0) {  // exactly 2 commutative adds per slot (wn=0,1): deterministic
        const int li = wm * 32 + 16 * mi + hb8 + v;
        atomicAdd(&rsx_s[li], r1);
        atomicAdd(&rsy_s[li], r2);
      }
    }
  }
#pragma unroll
  for (int off = 1; off < 32; off <<= 1) s2l += __shfl_xor(s2l, off, 32);
  if (lane == 0) s2_s[wave] = s2l;
  __syncthreads();
  if (threadIdx.x == 0) {
    float s = 0.f;
#pragma unroll
    for (int w = 0; w < 8; ++w) s += s2_s[w];
    S2p[bj * GRID_I + bi] = s;  // fixed slot, no global atomics
  }
  if (threadIdx.x < BM) {
    rsXp[(size_t)bj * HSIC_N + bi * BM + threadIdx.x] = rsx_s[threadIdx.x];
    rsYp[(size_t)bj * HSIC_N + bi * BM + threadIdx.x] = rsy_s[threadIdx.x];
  }
}

// ---------------------------------------------------------------------------
// Kernel 3: deterministic combine. 1 block.
// ---------------------------------------------------------------------------
__global__ __launch_bounds__(256) void hsic_finalize(
    const float* __restrict__ S2p, const float* __restrict__ rsXp,
    const float* __restrict__ rsYp, float* __restrict__ out) {
  double s2 = 0.0, dot = 0.0, sx = 0.0, sy = 0.0;
  for (int b = threadIdx.x; b < NBLK; b += 256) s2 += (double)S2p[b];
  for (int i = threadIdx.x; i < HSIC_N; i += 256) {
    float rx = 0.f, ry = 0.f;
    for (int bj = 0; bj < GRID_J; ++bj) {
      rx += rsXp[(size_t)bj * HSIC_N + i];
      ry += rsYp[(size_t)bj * HSIC_N + i];
    }
    dot += (double)rx * (double)ry;
    sx += (double)rx;
    sy += (double)ry;
  }
  __shared__ double red[256];
  auto block_reduce = [&](double v) -> double {
    red[threadIdx.x] = v;
    __syncthreads();
    for (int st = 128; st > 0; st >>= 1) {
      if (threadIdx.x < st) red[threadIdx.x] += red[threadIdx.x + st];
      __syncthreads();
    }
    double r = red[0];
    __syncthreads();
    return r;
  };
  s2 = block_reduce(s2);
  dot = block_reduce(dot);
  sx = block_reduce(sx);
  sy = block_reduce(sy);
  if (threadIdx.x == 0) {
    const double n = (double)HSIC_N;
    const double h = (s2 - (2.0 / n) * dot + (sx * sy) / (n * n)) /
                     ((n - 1.0) * (n - 1.0));
    out[0] = (float)h;
  }
}

// ---------------------------------------------------------------------------
extern "C" void kernel_launch(void* const* d_in, const int* in_sizes, int n_in,
                              void* d_out, int out_size, void* d_ws, size_t ws_size,
                              hipStream_t stream) {
  (void)in_sizes; (void)n_in; (void)out_size; (void)ws_size;
  const float* X = (const float*)d_in[0];
  const float* Y = (const float*)d_in[1];
  float* out = (float*)d_out;
  char* ws = (char*)d_ws;

  const size_t elems = (size_t)HSIC_N * HSIC_D;  // 2,097,152
  unsigned short* Xhi = (unsigned short*)(ws + 0 * elems);
  unsigned short* Xlo = (unsigned short*)(ws + 2 * elems);
  unsigned short* Yhi = (unsigned short*)(ws + 4 * elems);
  unsigned short* Ylo = (unsigned short*)(ws + 6 * elems);
  float* sqX  = (float*)(ws + 8 * elems);
  float* sqY  = sqX + HSIC_N;
  float* S2p  = sqY + HSIC_N;
  float* rsXp = S2p + NBLK;
  float* rsYp = rsXp + (size_t)GRID_J * HSIC_N;
  // total workspace: 16 MiB (bf16 splits) + ~2.1 MiB (partials)

  hsic_convert<<<dim3(HSIC_N, 2), 256, 0, stream>>>(X, Y, Xhi, Xlo, Yhi, Ylo,
                                                    sqX, sqY);
  hsic_main<<<dim3(GRID_I, GRID_J), 256, 0, stream>>>(Xhi, Xlo, Yhi, Ylo,
                                                      sqX, sqY, S2p, rsXp, rsYp);
  hsic_finalize<<<1, 256, 0, stream>>>(S2p, rsXp, rsYp, out);
}
